// EmbeddingLoss_13168369729809
// MI455X (gfx1250) — matrix operand order repaired and checked by reading, not compiled, not run on hardware
//
#include <hip/hip_runtime.h>
#include <hip/hip_bf16.h>
#include <math.h>

// Problem constants (from reference)
#define N_CHARS 16384
#define EMBED 256
#define TILE 16
#define KCHUNKS (EMBED / 32)     // 8 bf16 WMMA 16x16x32 K-chunks
#define ROWS_PER_BLOCK 128       // 8 waves * 16 rows
#define BSTRIDE 264              // 256 + 8 halves pad -> conflict-free b128 reads
#define NTILES (N_CHARS / TILE)  // 1024 column tiles
#define D2_BIG 3.0e38f           // pushes diagonal out of the d^2 min

typedef __attribute__((ext_vector_type(16))) __bf16 v16bf;
typedef __attribute__((ext_vector_type(8)))  __bf16 v8bf;
typedef __attribute__((ext_vector_type(8)))  float  v8f;
typedef __attribute__((ext_vector_type(4)))  int    v4i;

typedef __attribute__((address_space(1))) v4i gas_v4i;   // global
typedef __attribute__((address_space(3))) v4i las_v4i;   // LDS

#if __has_builtin(__builtin_amdgcn_global_load_async_to_lds_b128) && \
    __has_builtin(__builtin_amdgcn_s_wait_asynccnt)
#define HAS_ASYNC_LDS 1
#else
#define HAS_ASYNC_LDS 0
#endif

static __device__ __forceinline__ v16bf joinbf(v8bf a, v8bf b) {
    return __builtin_shufflevector(a, b, 0, 1, 2, 3, 4, 5, 6, 7,
                                          8, 9, 10, 11, 12, 13, 14, 15);
}

#if HAS_ASYNC_LDS
// One lane-wide 16-byte async copy global -> LDS (tracked by ASYNCcnt).
static __device__ __forceinline__ void async16(const __bf16* g, __bf16* l) {
    void* gv = (void*)g;   // strip const, then addrspace+type cast
    void* lv = (void*)l;
    __builtin_amdgcn_global_load_async_to_lds_b128(
        (gas_v4i*)gv, (las_v4i*)lv, /*offset=*/0, /*cpol=*/0);
}
#endif

// ---------------------------------------------------------------------------
// Kernel 0: fp32 -> (bf16 hi, bf16 lo) split, row-major.  W ~= hi + lo with
// residual ~2^-16 relative, so hi*hi + hi*lo + lo*hi recovers ~fp32 gram.
// ---------------------------------------------------------------------------
__global__ __launch_bounds__(256) void cvt_kernel(const float* __restrict__ W,
                                                  __bf16* __restrict__ Whi,
                                                  __bf16* __restrict__ Wlo) {
    int row = blockIdx.x * blockDim.x + threadIdx.x;
    const float4* p  = (const float4*)(W + (size_t)row * EMBED);
    __bf16*       ph = Whi + (size_t)row * EMBED;
    __bf16*       pl = Wlo + (size_t)row * EMBED;
#pragma unroll 4
    for (int i = 0; i < EMBED / 4; ++i) {
        float4 v = p[i];
        __bf16 h0 = (__bf16)v.x, h1 = (__bf16)v.y, h2 = (__bf16)v.z, h3 = (__bf16)v.w;
        ph[4 * i + 0] = h0; ph[4 * i + 1] = h1; ph[4 * i + 2] = h2; ph[4 * i + 3] = h3;
        pl[4 * i + 0] = (__bf16)(v.x - (float)h0);
        pl[4 * i + 1] = (__bf16)(v.y - (float)h1);
        pl[4 * i + 2] = (__bf16)(v.z - (float)h2);
        pl[4 * i + 3] = (__bf16)(v.w - (float)h3);
    }
}

// ---------------------------------------------------------------------------
// Kernel 1: per-row squared norms in full fp32  sq[i] = sum_k W[i][k]^2
// ---------------------------------------------------------------------------
__global__ __launch_bounds__(256) void sq_kernel(const float* __restrict__ W,
                                                 float* __restrict__ sq) {
    int row = blockIdx.x * blockDim.x + threadIdx.x;
    const float4* p = (const float4*)(W + (size_t)row * EMBED);
    float s = 0.0f;
#pragma unroll
    for (int i = 0; i < EMBED / 4; ++i) {
        float4 v = p[i];
        s += v.x * v.x + v.y * v.y + v.z * v.z + v.w * v.w;
    }
    sq[row] = s;
}

// ---------------------------------------------------------------------------
// Kernel 2: fused Gram (bf16 WMMA 16x16x32, 2-term split) -> d^2 -> per-row
// running min.  Double-buffered LDS column tiles, prefetched with async
// global->LDS copies (ASYNCcnt) while WMMAs run on the other buffer.
// ---------------------------------------------------------------------------
__global__ __launch_bounds__(256) void minDistKernel(const __bf16* __restrict__ Whi,
                                                     const __bf16* __restrict__ Wlo,
                                                     const float* __restrict__ sq,
                                                     float* __restrict__ minOut) {
    __shared__ __bf16 ldsHi[2][TILE * BSTRIDE];
    __shared__ __bf16 ldsLo[2][TILE * BSTRIDE];

    const int tid    = threadIdx.x;
    const int wave   = tid >> 5;
    const int lane   = tid & 31;
    const int lane15 = lane & 15;
    const int half   = lane >> 4;     // 0: lanes 0-15, 1: lanes 16-31
    const int m      = blockIdx.x * ROWS_PER_BLOCK + wave * TILE;
    const int arow   = m + lane15;
    const int mtile  = m >> 4;        // the only column tile containing diagonal

    // Stage one 16-column tile (hi+lo) into LDS buffer `buf`.
    auto stage = [&](int jj, int buf) {
        int r  = tid >> 4;            // tile row 0..15
        int c0 = (tid & 15) * 16;     // 16 halves = 32 B per thread per matrix
        const __bf16* gh = Whi + (size_t)(jj * TILE + r) * EMBED + c0;
        const __bf16* gl = Wlo + (size_t)(jj * TILE + r) * EMBED + c0;
        __bf16* lh = &ldsHi[buf][r * BSTRIDE + c0];
        __bf16* ll = &ldsLo[buf][r * BSTRIDE + c0];
#if HAS_ASYNC_LDS
        async16(gh, lh);  async16(gh + 8, lh + 8);
        async16(gl, ll);  async16(gl + 8, ll + 8);
#else
        uint4 h0 = ((const uint4*)gh)[0], h1 = ((const uint4*)gh)[1];
        uint4 l0 = ((const uint4*)gl)[0], l1 = ((const uint4*)gl)[1];
        ((uint4*)lh)[0] = h0; ((uint4*)lh)[1] = h1;
        ((uint4*)ll)[0] = l0; ((uint4*)ll)[1] = l1;
#endif
    };

    // A fragments (ISA 16-bit 16x32 layout): per lane, elements 0-7 hold
    // K = 8*half+{0..7}, elements 8-15 hold K = 16+8*half+{0..7}.
    v16bf afragHi[KCHUNKS];
    v16bf afragLo[KCHUNKS];
#pragma unroll
    for (int kc = 0; kc < KCHUNKS; ++kc) {
        const __bf16* baseH = Whi + (size_t)arow * EMBED + kc * 32 + half * 8;
        const __bf16* baseL = Wlo + (size_t)arow * EMBED + kc * 32 + half * 8;
        afragHi[kc] = joinbf(*(const v8bf*)(baseH), *(const v8bf*)(baseH + 16));
        afragLo[kc] = joinbf(*(const v8bf*)(baseL), *(const v8bf*)(baseL + 16));
    }

    // C/D layout: VGPR r on lanes 0-15 is row m+r, on lanes 16-31 row m+r+8.
    float sqrow[8];
#pragma unroll
    for (int r = 0; r < 8; ++r) sqrow[r] = sq[m + r + half * 8];

    float runmin2[8];                 // running min of d^2 (sqrt once at end)
#pragma unroll
    for (int r = 0; r < 8; ++r) runmin2[r] = D2_BIG;

    stage(0, 0);                      // prefetch first tile

    for (int j = 0; j < NTILES; ++j) {
        const int cur = j & 1;
#if HAS_ASYNC_LDS
        __builtin_amdgcn_s_wait_asynccnt(0);   // my copies into `cur` done
#endif
        __syncthreads();              // all waves' copies visible; `nxt` free
        if (j + 1 < NTILES) stage(j + 1, cur ^ 1);

        float sqc = sq[j * TILE + lane15];     // L2-resident, 1 load/lane/tile

        // Two accumulators (even/odd chunks) to break the WMMA RAW chain.
        v8f acc0 = {0.f, 0.f, 0.f, 0.f, 0.f, 0.f, 0.f, 0.f};
        v8f acc1 = {0.f, 0.f, 0.f, 0.f, 0.f, 0.f, 0.f, 0.f};
#pragma unroll
        for (int kc = 0; kc < KCHUNKS; ++kc) {
            // B fragment (32x16 bf16): per lane, elements 0-7 hold K = 8*half+{0..7}, elements 8-15 hold K = 16+8*half+{0..7}.
            int bofs = kc * 32 + half * 8;
            const __bf16* bh = &ldsHi[cur][lane15 * BSTRIDE + bofs];
            const __bf16* bl = &ldsLo[cur][lane15 * BSTRIDE + bofs];
            v16bf bhv = joinbf(*(const v8bf*)(bh), *(const v8bf*)(bh + 16));
            v16bf blv = joinbf(*(const v8bf*)(bl), *(const v8bf*)(bl + 16));
            if ((kc & 1) == 0) {
                acc0 = __builtin_amdgcn_wmma_f32_16x16x32_bf16(false, afragHi[kc], false, bhv,
                                                               (short)0, acc0, false, false);
                acc0 = __builtin_amdgcn_wmma_f32_16x16x32_bf16(false, afragHi[kc], false, blv,
                                                               (short)0, acc0, false, false);
                acc0 = __builtin_amdgcn_wmma_f32_16x16x32_bf16(false, afragLo[kc], false, bhv,
                                                               (short)0, acc0, false, false);
            } else {
                acc1 = __builtin_amdgcn_wmma_f32_16x16x32_bf16(false, afragHi[kc], false, bhv,
                                                               (short)0, acc1, false, false);
                acc1 = __builtin_amdgcn_wmma_f32_16x16x32_bf16(false, afragHi[kc], false, blv,
                                                               (short)0, acc1, false, false);
                acc1 = __builtin_amdgcn_wmma_f32_16x16x32_bf16(false, afragLo[kc], false, bhv,
                                                               (short)0, acc1, false, false);
            }
        }

        if (j == mtile) {             // uniform branch: only tile with diagonal
#pragma unroll
            for (int r = 0; r < 8; ++r) {
                float d2 = (sqrow[r] + sqc) - 2.0f * (acc0[r] + acc1[r]);
                if (r + half * 8 == lane15) d2 = D2_BIG;   // row == col
                runmin2[r] = fminf(runmin2[r], d2);
            }
        } else {
#pragma unroll
            for (int r = 0; r < 8; ++r) {
                float d2 = (sqrow[r] + sqc) - 2.0f * (acc0[r] + acc1[r]);
                runmin2[r] = fminf(runmin2[r], d2);
            }
        }
    }

    // Min of d^2 across the 16 lanes sharing each row, then one sqrt per row.
#pragma unroll
    for (int r = 0; r < 8; ++r) {
        float v = runmin2[r];
#pragma unroll
        for (int off = 8; off >= 1; off >>= 1)
            v = fminf(v, __shfl_xor(v, off, 16));
        if (lane15 == 0) minOut[m + r + half * 8] = sqrtf(fmaxf(v, 0.0f));
    }
}

// ---------------------------------------------------------------------------
// Kernel 3: mean of mins, masked mean -> out = {loss, mean}
// ---------------------------------------------------------------------------
__global__ __launch_bounds__(256) void finalizeKernel(const float* __restrict__ minDist,
                                                      float* __restrict__ out) {
    __shared__ float red[256];
    __shared__ float s_mean;
    int tid = threadIdx.x;

    float s = 0.0f;
    for (int i = tid; i < N_CHARS; i += 256) s += minDist[i];
    red[tid] = s;
    __syncthreads();
    for (int o = 128; o > 0; o >>= 1) {
        if (tid < o) red[tid] += red[tid + o];
        __syncthreads();
    }
    if (tid == 0) s_mean = red[0] / (float)N_CHARS;
    __syncthreads();
    float mean = s_mean;

    float ms = 0.0f;
    for (int i = tid; i < N_CHARS; i += 256) {
        float v = minDist[i];
        if (v <= mean) ms += v;   // where(min > mean, 0, min)
    }
    red[tid] = ms;
    __syncthreads();
    for (int o = 128; o > 0; o >>= 1) {
        if (tid < o) red[tid] += red[tid + o];
        __syncthreads();
    }
    if (tid == 0) {
        out[0] = -red[0] / (float)N_CHARS;  // loss
        out[1] = mean;                       // mean_distance
    }
}

// ---------------------------------------------------------------------------
extern "C" void kernel_launch(void* const* d_in, const int* in_sizes, int n_in,
                              void* d_out, int out_size, void* d_ws, size_t ws_size,
                              hipStream_t stream) {
    const float* W   = (const float*)d_in[0];
    float*       out = (float*)d_out;

    // Workspace layout: sq (64KB) | minDist (64KB) | Whi (8MB) | Wlo (8MB)
    float*  sq      = (float*)d_ws;
    float*  minDist = (float*)d_ws + N_CHARS;
    __bf16* Whi     = (__bf16*)((char*)d_ws + 2 * N_CHARS * sizeof(float));
    __bf16* Wlo     = Whi + (size_t)N_CHARS * EMBED;

    cvt_kernel<<<N_CHARS / 256, 256, 0, stream>>>(W, Whi, Wlo);
    sq_kernel<<<N_CHARS / 256, 256, 0, stream>>>(W, sq);
    minDistKernel<<<N_CHARS / ROWS_PER_BLOCK, 256, 0, stream>>>(Whi, Wlo, sq, minDist);
    finalizeKernel<<<1, 256, 0, stream>>>(minDist, out);
}
